// GCN_266287972964
// MI455X (gfx1250) — matrix operand order repaired and checked by reading, not compiled, not run on hardware
//
#include <hip/hip_runtime.h>
#include <stdint.h>

#define NN 50000
#define NE 800000
#define F_IN 512
#define F_HID 128
#define F_OUT 40
#define F_OUT_PAD 48

typedef __attribute__((ext_vector_type(16))) __bf16 v16bf;
typedef __attribute__((ext_vector_type(8)))  float  v8f;

struct alignas(16) U4 { unsigned int x, y, z, w; };
union AV { U4 q[2]; v16bf v; };

__device__ __forceinline__ unsigned short f2bf(float x) {
    unsigned u = __float_as_uint(x);
    unsigned r = u + 0x7FFFu + ((u >> 16) & 1u);   // round-to-nearest-even
    return (unsigned short)(r >> 16);
}

// ---------------- degree / norm ----------------
__global__ void k_degree(const int* __restrict__ src, const int* __restrict__ dst,
                         float* __restrict__ deg_s, float* __restrict__ deg_d, int E) {
    int e = blockIdx.x * blockDim.x + threadIdx.x;
    if (e < E) {
        atomicAdd(&deg_s[src[e]], 1.0f);
        atomicAdd(&deg_d[dst[e]], 1.0f);
    }
}

__global__ void k_invsqrt(float* __restrict__ a, float* __restrict__ b, int n) {
    int i = blockIdx.x * blockDim.x + threadIdx.x;
    if (i < n) {
        float x = a[i]; a[i] = x > 0.0f ? rsqrtf(x) : 0.0f;
        float y = b[i]; b[i] = y > 0.0f ? rsqrtf(y) : 0.0f;
    }
}

// ---------------- scale + cast x to bf16 ----------------
__global__ void k_scale_x(const float* __restrict__ x, const float* __restrict__ ns,
                          unsigned short* __restrict__ xs, int total) {
    int t = blockIdx.x * blockDim.x + threadIdx.x;
    if (t < total) {
        int i = t >> 9;                           // F_IN = 512
        xs[t] = f2bf(x[t] * ns[i]);
    }
}

// ---- transpose+cast W [K x Nc] row-major -> Wt [Npad x K] (B column-major), zero pad ----
__global__ void k_wt(const float* __restrict__ W, unsigned short* __restrict__ Wt,
                     int K, int Nc, int Npad) {
    int t = blockIdx.x * blockDim.x + threadIdx.x;
    if (t < Npad * K) {
        int n = t / K, k = t - n * K;
        Wt[t] = (n < Nc) ? f2bf(W[k * Nc + n]) : (unsigned short)0;
    }
}

// ------------- bf16 WMMA GEMM, register-blocked over all N tiles -------------
// A row-major bf16 [M x K], Bt = B column-major bf16 ([NT*16 x K] row-major), C f32.
// One wave computes a 16 x (NT*16) slab: A chunk loaded once, reused by NT WMMAs.
// grid.x = M/16.
template <int K, int NT>
__global__ void __launch_bounds__(32)
k_gemm_bf16(const unsigned short* __restrict__ A,
            const unsigned short* __restrict__ Bt,
            float* __restrict__ C, int ldc) {
    const int lane = threadIdx.x & 31;
    const int half = lane >> 4;
    const int m16  = lane & 15;

    const unsigned short* rowA  = A  + (size_t)(blockIdx.x * 16 + m16) * K;
    const unsigned short* rowB0 = Bt + (size_t)m16 * K;

    v8f acc[NT];
#pragma unroll
    for (int t = 0; t < NT; ++t) acc[t] = {};

#pragma unroll
    for (int kb = 0; kb < K; kb += 32) {
        AV a;
        // A 16x32 bf16: lanes 0-15 = M rows; halves K 0-15/16-31 (contiguous, matches B)
        a.q[0] = *reinterpret_cast<const U4*>(rowA + kb + 16 * half);       // elems 0..7
        a.q[1] = *reinterpret_cast<const U4*>(rowA + kb + 16 * half + 8);   // elems 8..15
#pragma unroll
        for (int t = 0; t < NT; ++t) {
            AV b;
            // B 32x16 bf16 (column-major source): lane = N column; halves K 0-15/16-31
            const unsigned short* rb = rowB0 + (size_t)t * 16 * K + kb + 16 * half;
            b.q[0] = *reinterpret_cast<const U4*>(rb);
            b.q[1] = *reinterpret_cast<const U4*>(rb + 8);
            acc[t] = __builtin_amdgcn_wmma_f32_16x16x32_bf16(false, a.v, false, b.v,
                                                             (short)0, acc[t], false, false);
        }
    }

    float* outp = C + (size_t)(blockIdx.x * 16 + 8 * half) * ldc + m16;
#pragma unroll
    for (int t = 0; t < NT; ++t)
#pragma unroll
        for (int r = 0; r < 8; ++r)
            outp[(size_t)r * ldc + t * 16] = acc[t][r];
}

// ---------------- edge scatter: agg[dst] += h[src] ----------------
__global__ void k_scatter(const int* __restrict__ src, const int* __restrict__ dst,
                          const float* __restrict__ h, float* __restrict__ agg,
                          int E, int feat, int src_ld, int dst_ld) {
    int t = blockIdx.x * blockDim.x + threadIdx.x;
    if (t < E * feat) {
        int e = t / feat, f = t - e * feat;
        atomicAdd(&agg[(size_t)dst[e] * dst_ld + f], h[(size_t)src[e] * src_ld + f]);
    }
}

// ---- layer-1 epilogue: relu(agg*nd + b1) * ns  -> bf16 for GEMM2 ----
__global__ void k_post1(const float* __restrict__ agg, const float* __restrict__ nd,
                        const float* __restrict__ ns, const float* __restrict__ b1,
                        unsigned short* __restrict__ hs, int total) {
    int t = blockIdx.x * blockDim.x + threadIdx.x;
    if (t < total) {
        int i = t >> 7, f = t & 127;              // F_HID = 128
        float v = agg[t] * nd[i] + b1[f];
        v = v > 0.0f ? v : 0.0f;
        hs[t] = f2bf(v * ns[i]);
    }
}

// ---- final epilogue: out = agg*nd + b2 ----
__global__ void k_out(const float* __restrict__ agg, const float* __restrict__ nd,
                      const float* __restrict__ b2, float* __restrict__ out, int total) {
    int t = blockIdx.x * blockDim.x + threadIdx.x;
    if (t < total) {
        int i = t / F_OUT, f = t - i * F_OUT;
        out[t] = agg[t] * nd[i] + b2[f];
    }
}

extern "C" void kernel_launch(void* const* d_in, const int* in_sizes, int n_in,
                              void* d_out, int out_size, void* d_ws, size_t ws_size,
                              hipStream_t stream) {
    const float* x   = (const float*)d_in[0];
    const float* W1  = (const float*)d_in[1];
    const float* b1  = (const float*)d_in[2];
    const float* W2  = (const float*)d_in[3];
    const float* b2  = (const float*)d_in[4];
    const int*   src = (const int*)d_in[5];
    const int*   dst = (const int*)d_in[6];
    float* out = (float*)d_out;

    // ---- carve workspace ----
    char* p = (char*)d_ws;
    auto carve = [&](size_t bytes) { void* r = p; p += (bytes + 255) & ~(size_t)255; return r; };
    float*          norm_s = (float*)carve((size_t)NN * 4);
    float*          norm_d = (float*)carve((size_t)NN * 4);
    unsigned short* xs     = (unsigned short*)carve((size_t)NN * F_IN * 2);
    unsigned short* Wt1    = (unsigned short*)carve((size_t)F_HID * F_IN * 2);
    unsigned short* Wt2    = (unsigned short*)carve((size_t)F_OUT_PAD * F_HID * 2);
    float*          h1     = (float*)carve((size_t)NN * F_HID * 4);
    float*          agg1   = (float*)carve((size_t)NN * F_HID * 4);
    unsigned short* hs     = (unsigned short*)carve((size_t)NN * F_HID * 2);
    float*          t2     = (float*)carve((size_t)NN * F_OUT_PAD * 4);
    float*          agg2   = (float*)carve((size_t)NN * F_OUT * 4);
    (void)ws_size; (void)in_sizes; (void)n_in; (void)out_size;

    const int B = 256;
    auto blocks = [](long long n, int b) { return (unsigned)((n + b - 1) / b); };

    // 1) degrees -> inv-sqrt norms
    hipMemsetAsync(norm_s, 0, (size_t)NN * 4, stream);
    hipMemsetAsync(norm_d, 0, (size_t)NN * 4, stream);
    k_degree<<<blocks(NE, B), B, 0, stream>>>(src, dst, norm_s, norm_d, NE);
    k_invsqrt<<<blocks(NN, B), B, 0, stream>>>(norm_s, norm_d, NN);

    // 2) prepare bf16 operands
    k_scale_x<<<blocks((long long)NN * F_IN, B), B, 0, stream>>>(x, norm_s, xs, NN * F_IN);
    k_wt<<<blocks((long long)F_HID * F_IN, B), B, 0, stream>>>(W1, Wt1, F_IN, F_HID, F_HID);
    k_wt<<<blocks((long long)F_OUT_PAD * F_HID, B), B, 0, stream>>>(W2, Wt2, F_HID, F_OUT, F_OUT_PAD);

    // 3) GEMM1: h1[N,128] = xs @ W1   (WMMA bf16, f32 accum; 16x128 slab per wave)
    k_gemm_bf16<F_IN, F_HID / 16><<<dim3(NN / 16), 32, 0, stream>>>(xs, Wt1, h1, F_HID);

    // 4) layer-1 aggregation + epilogue
    hipMemsetAsync(agg1, 0, (size_t)NN * F_HID * 4, stream);
    k_scatter<<<blocks((long long)NE * F_HID, B), B, 0, stream>>>(src, dst, h1, agg1,
                                                                  NE, F_HID, F_HID, F_HID);
    k_post1<<<blocks((long long)NN * F_HID, B), B, 0, stream>>>(agg1, norm_d, norm_s, b1,
                                                                hs, NN * F_HID);

    // 5) GEMM2: t2[N,48] = hs @ W2pad   (WMMA bf16; 16x48 slab per wave)
    k_gemm_bf16<F_HID, F_OUT_PAD / 16><<<dim3(NN / 16), 32, 0, stream>>>(hs, Wt2, t2, F_OUT_PAD);

    // 6) layer-2 aggregation + output
    hipMemsetAsync(agg2, 0, (size_t)NN * F_OUT * 4, stream);
    k_scatter<<<blocks((long long)NE * F_OUT, B), B, 0, stream>>>(src, dst, t2, agg2,
                                                                  NE, F_OUT, F_OUT_PAD, F_OUT);
    k_out<<<blocks((long long)NN * F_OUT, B), B, 0, stream>>>(agg2, norm_d, b2, out, NN * F_OUT);
}
